// MultiAttentionHead_36799279792243
// MI455X (gfx1250) — matrix operand order repaired and checked by reading, not compiled, not run on hardware
//
#include <hip/hip_runtime.h>
#include <hip/hip_bf16.h>

// ---------------------------------------------------------------------------
// MHA for MI455X (gfx1250): bf16 WMMA everywhere, flash-style attention.
// B=4, S=2048, D=1024, H=16, DH=64
// ---------------------------------------------------------------------------

#define Bsz 4
#define Ssz 2048
#define Dsz 1024
#define Hsz 16
#define DHsz 64

typedef __bf16 bf16;
typedef __attribute__((ext_vector_type(8)))  __bf16 v8bf;
typedef __attribute__((ext_vector_type(16))) __bf16 v16bf;
typedef __attribute__((ext_vector_type(8)))  float  v8f;

__device__ __forceinline__ v8f zero_v8f() {
  v8f z;
#pragma unroll
  for (int i = 0; i < 8; ++i) z[i] = 0.0f;
  return z;
}

// A-matrix fragment (16x32 bf16, MxK): per ISA table, lane half 0 holds
// K=[k0..k0+7] and [k0+16..k0+23]; half 1 holds [k0+8..15] and [k0+24..31].
__device__ __forceinline__ v16bf frag_A(const bf16* rowptr, int k0, int half) {
  const v8bf* p0 = reinterpret_cast<const v8bf*>(rowptr + k0 + half * 8);
  const v8bf* p1 = reinterpret_cast<const v8bf*>(rowptr + k0 + half * 8 + 16);
  v8bf a = *p0, b = *p1;
  return __builtin_shufflevector(a, b, 0, 1, 2, 3, 4, 5, 6, 7,
                                 8, 9, 10, 11, 12, 13, 14, 15);
}

// B-matrix fragment (32x16 bf16, KxN): lane = column, half 0 holds K=[k0..k0+7]
// and [k0+16..k0+23]; half 1 holds [k0+8..15] and [k0+24..31] (matches frag_A).
__device__ __forceinline__ v16bf frag_B(const bf16* rowptr, int k0, int half) {
  const v8bf* p0 = reinterpret_cast<const v8bf*>(rowptr + k0 + half * 8);
  const v8bf* p1 = reinterpret_cast<const v8bf*>(rowptr + k0 + half * 8 + 16);
  v8bf a = *p0, b = *p1;
  return __builtin_shufflevector(a, b, 0, 1, 2, 3, 4, 5, 6, 7,
                                 8, 9, 10, 11, 12, 13, 14, 15);
}

__device__ __forceinline__ v8f wmma_bf16(v16bf a, v16bf b, v8f c) {
  return __builtin_amdgcn_wmma_f32_16x16x32_bf16(
      false, a, false, b, (short)0, c, false, false);
}

// ---------------------------------------------------------------------------
// f32 -> bf16 conversion (grid-stride)
// ---------------------------------------------------------------------------
__global__ void cvt_f32_bf16(const float* __restrict__ src,
                             bf16* __restrict__ dst, int n) {
  int i = blockIdx.x * blockDim.x + threadIdx.x;
  int stride = gridDim.x * blockDim.x;
  for (; i < n; i += stride) dst[i] = (bf16)src[i];
}

// ---------------------------------------------------------------------------
// GEMM: C[M=8192, N=1024] = A[M,1024] @ W[N,1024]^T + bias[N]
// mode 0: store bf16 as Q/K layout [B,H,S,DH]
// mode 1: store bf16 as V^T layout [B,H,DH,S]
// mode 2: store f32 row-major [M,N]
// Block: 256 thr = 8 waves (2x4), wave tile 64x64, block tile 128x256,
// K-chunks of 32 -> 16 v_wmma vs 16 b128 loads per chunk (~65 FLOP/L2-byte).
// ---------------------------------------------------------------------------
__global__ __launch_bounds__(256) void gemm_bf16_k(
    const bf16* __restrict__ A, const bf16* __restrict__ W,
    const float* __restrict__ bias, void* __restrict__ out, int mode) {
  const int tid = threadIdx.x;
  const int wave = tid >> 5;
  const int lane = tid & 31;
  const int lrow = lane & 15;
  const int half = lane >> 4;
  const int wm = wave >> 2;  // 0..1
  const int wn = wave & 3;   // 0..3
  const int m0 = blockIdx.y * 128 + wm * 64;
  const int n0 = blockIdx.x * 256 + wn * 64;

  const bf16* Arow[4];
#pragma unroll
  for (int i = 0; i < 4; ++i)
    Arow[i] = A + (size_t)(m0 + 16 * i + lrow) * Dsz;
  const bf16* Wrow[4];
#pragma unroll
  for (int j = 0; j < 4; ++j)
    Wrow[j] = W + (size_t)(n0 + 16 * j + lrow) * Dsz;

  v8f acc[4][4];
#pragma unroll
  for (int i = 0; i < 4; ++i)
#pragma unroll
    for (int j = 0; j < 4; ++j) acc[i][j] = zero_v8f();

  for (int k0 = 0; k0 < Dsz; k0 += 32) {
    // prefetch next chunk (near-scope)
    if (k0 + 32 < Dsz) {
      __builtin_prefetch(Arow[0] + k0 + 32, 0, 3);
      __builtin_prefetch(Wrow[0] + k0 + 32, 0, 3);
    }
    v16bf af[4];
#pragma unroll
    for (int i = 0; i < 4; ++i) af[i] = frag_A(Arow[i], k0, half);
    v16bf bfm[4];
#pragma unroll
    for (int j = 0; j < 4; ++j) bfm[j] = frag_B(Wrow[j], k0, half);
#pragma unroll
    for (int i = 0; i < 4; ++i)
#pragma unroll
      for (int j = 0; j < 4; ++j) acc[i][j] = wmma_bf16(af[i], bfm[j], acc[i][j]);
  }

#pragma unroll
  for (int i = 0; i < 4; ++i) {
#pragma unroll
    for (int j = 0; j < 4; ++j) {
      const int n = n0 + 16 * j + lrow;
      const float bb = bias[n];
#pragma unroll
      for (int r = 0; r < 8; ++r) {
        const int m = m0 + 16 * i + r + 8 * half;
        const float v = acc[i][j][r] + bb;
        if (mode == 2) {
          ((float*)out)[(size_t)m * Dsz + n] = v;
        } else {
          const int b = m >> 11, s = m & (Ssz - 1);
          const int h = n >> 6, dh = n & (DHsz - 1);
          bf16* o = (bf16*)out;
          if (mode == 0)  // [B,H,S,DH]
            o[(((size_t)(b * Hsz + h)) * Ssz + s) * DHsz + dh] = (bf16)v;
          else            // [B,H,DH,S]  (V transposed)
            o[(((size_t)(b * Hsz + h)) * DHsz + dh) * Ssz + s] = (bf16)v;
        }
      }
    }
  }
}

// ---------------------------------------------------------------------------
// Flash attention. Block = one (b,h) x 64 query rows; 4 waves x 16 rows.
// Per 32-key chunk: 4 WMMA for QK^T, online softmax (shfl_xor within the
// 16-lane half), P staged bf16 in LDS (padded rows), 4 WMMA for P.V.
// ctx output: [B,S,D] bf16.
// ---------------------------------------------------------------------------
#define PSTRIDE 40  // 80B row stride: 16B aligned, conflict-free mod 256B

__global__ __launch_bounds__(128) void attn_k(
    const bf16* __restrict__ Q, const bf16* __restrict__ K,
    const bf16* __restrict__ Vt, const int* __restrict__ lens,
    bf16* __restrict__ ctx) {
  const int bh = blockIdx.x;        // b*16 + h
  const int b = bh >> 4;
  const int h = bh & 15;
  const int L = lens[b];

  const int wave = threadIdx.x >> 5;
  const int lane = threadIdx.x & 31;
  const int lrow = lane & 15;
  const int half = lane >> 4;
  const int qbase = blockIdx.y * 64 + wave * 16;
  const int qrow = qbase + lrow;

  __shared__ bf16 P[4][16][PSTRIDE];

  const bf16* Qrow = Q + ((size_t)bh * Ssz + qrow) * DHsz;
  const v16bf qa0 = frag_A(Qrow, 0, half);
  const v16bf qa1 = frag_A(Qrow, 32, half);

  v8f acc[4];
#pragma unroll
  for (int j = 0; j < 4; ++j) acc[j] = zero_v8f();
  float mst[8], lst[8];
#pragma unroll
  for (int r = 0; r < 8; ++r) { mst[r] = -1e30f; lst[r] = 0.0f; }

  for (int kc = 0; kc < Ssz; kc += 32) {
    // ---- scores: two 16x16 tiles over keys [kc..kc+31]
    v8f t[2];
#pragma unroll
    for (int ti = 0; ti < 2; ++ti) {
      const int key = kc + 16 * ti + lrow;
      const bf16* Krow = K + ((size_t)bh * Ssz + key) * DHsz;
      v16bf kb0 = frag_B(Krow, 0, half);
      v16bf kb1 = frag_B(Krow, 32, half);
      v8f s = zero_v8f();
      s = wmma_bf16(qa0, kb0, s);
      s = wmma_bf16(qa1, kb1, s);
      t[ti] = s;
    }
    // ---- scale + ragged mask (matches reference: masked -> exactly -1e9)
#pragma unroll
    for (int ti = 0; ti < 2; ++ti) {
      const int key = kc + 16 * ti + lrow;
      const bool kv = key < L;
#pragma unroll
      for (int r = 0; r < 8; ++r) {
        const int qp = qbase + r + 8 * half;
        t[ti][r] = (kv && (qp < L)) ? t[ti][r] * 0.125f : -1e9f;
      }
    }
    // ---- online softmax: row reductions across the 16-lane half
    float mc[8];
#pragma unroll
    for (int r = 0; r < 8; ++r) mc[r] = fmaxf(t[0][r], t[1][r]);
#pragma unroll
    for (int off = 1; off < 16; off <<= 1)
#pragma unroll
      for (int r = 0; r < 8; ++r)
        mc[r] = fmaxf(mc[r], __shfl_xor(mc[r], off, 32));
    float al[8];
#pragma unroll
    for (int r = 0; r < 8; ++r) {
      const float mn = fmaxf(mst[r], mc[r]);
      al[r] = __expf(mst[r] - mn);
      mst[r] = mn;
    }
#pragma unroll
    for (int r = 0; r < 8; ++r) {
      t[0][r] = __expf(t[0][r] - mst[r]);
      t[1][r] = __expf(t[1][r] - mst[r]);
    }
    float rs[8];
#pragma unroll
    for (int r = 0; r < 8; ++r) rs[r] = t[0][r] + t[1][r];
#pragma unroll
    for (int off = 1; off < 16; off <<= 1)
#pragma unroll
      for (int r = 0; r < 8; ++r) rs[r] += __shfl_xor(rs[r], off, 32);
#pragma unroll
    for (int r = 0; r < 8; ++r) lst[r] = lst[r] * al[r] + rs[r];
    // ---- rescale accumulators
#pragma unroll
    for (int j = 0; j < 4; ++j)
#pragma unroll
      for (int r = 0; r < 8; ++r) acc[j][r] *= al[r];
    // ---- stage P (bf16) in LDS: C-layout -> row-major 16x32
#pragma unroll
    for (int r = 0; r < 8; ++r) {
      P[wave][r + 8 * half][lrow] = (bf16)t[0][r];
      P[wave][r + 8 * half][16 + lrow] = (bf16)t[1][r];
    }
    __syncthreads();  // all waves iterate identically; orders LDS w->r
    // ---- P @ V: A-frag of P from LDS, B-frag of V^T contiguous from global
    const v16bf pa = frag_A(&P[wave][lrow][0], 0, half);
#pragma unroll
    for (int j = 0; j < 4; ++j) {
      const bf16* Vrow = Vt + ((size_t)bh * DHsz + 16 * j + lrow) * Ssz;
      v16bf vb = frag_B(Vrow, kc, half);
      acc[j] = wmma_bf16(pa, vb, acc[j]);
    }
    __syncthreads();  // protect P against next iteration's writes
  }

  // ---- normalize and emit ctx[B,S,D] bf16
#pragma unroll
  for (int j = 0; j < 4; ++j) {
#pragma unroll
    for (int r = 0; r < 8; ++r) {
      const int qp = qbase + r + 8 * half;
      const int dcol = 16 * j + lrow;
      const float ov = acc[j][r] / lst[r];
      ctx[((size_t)b * Ssz + qp) * Dsz + h * DHsz + dcol] = (bf16)ov;
    }
  }
}

// ---------------------------------------------------------------------------
// Launch
// ---------------------------------------------------------------------------
extern "C" void kernel_launch(void* const* d_in, const int* in_sizes, int n_in,
                              void* d_out, int out_size, void* d_ws,
                              size_t ws_size, hipStream_t stream) {
  const float* x = (const float*)d_in[0];
  const int* lens = (const int*)d_in[1];
  const float* Wq = (const float*)d_in[2];
  const float* bq = (const float*)d_in[3];
  const float* Wk = (const float*)d_in[4];
  const float* bk = (const float*)d_in[5];
  const float* Wv = (const float*)d_in[6];
  const float* bv = (const float*)d_in[7];
  const float* Wo = (const float*)d_in[8];
  const float* bo = (const float*)d_in[9];
  float* out = (float*)d_out;

  const int XN = Bsz * Ssz * Dsz;  // 8388608
  const int WN = Dsz * Dsz;        // 1048576

  char* ws = (char*)d_ws;
  size_t off = 0;
  bf16* xb  = (bf16*)(ws + off); off += (size_t)XN * 2;  // 16 MB
  bf16* wqb = (bf16*)(ws + off); off += (size_t)WN * 2;  //  2 MB
  bf16* wkb = (bf16*)(ws + off); off += (size_t)WN * 2;
  bf16* wvb = (bf16*)(ws + off); off += (size_t)WN * 2;
  bf16* wob = (bf16*)(ws + off); off += (size_t)WN * 2;
  bf16* Qb  = (bf16*)(ws + off); off += (size_t)XN * 2;  // [B,H,S,DH]
  bf16* Kb  = (bf16*)(ws + off); off += (size_t)XN * 2;  // [B,H,S,DH]
  bf16* Vtb = (bf16*)(ws + off); off += (size_t)XN * 2;  // [B,H,DH,S]
  bf16* ctx = (bf16*)(ws + off); off += (size_t)XN * 2;  // [B,S,D]
  (void)ws_size;

  cvt_f32_bf16<<<2048, 256, 0, stream>>>(x, xb, XN);
  cvt_f32_bf16<<<512, 256, 0, stream>>>(Wq, wqb, WN);
  cvt_f32_bf16<<<512, 256, 0, stream>>>(Wk, wkb, WN);
  cvt_f32_bf16<<<512, 256, 0, stream>>>(Wv, wvb, WN);
  cvt_f32_bf16<<<512, 256, 0, stream>>>(Wo, wob, WN);

  dim3 ggrid(Dsz / 256, (Bsz * Ssz) / 128);  // (4, 64)
  gemm_bf16_k<<<ggrid, 256, 0, stream>>>(xb, wqb, bq, Qb, 0);
  gemm_bf16_k<<<ggrid, 256, 0, stream>>>(xb, wkb, bk, Kb, 0);
  gemm_bf16_k<<<ggrid, 256, 0, stream>>>(xb, wvb, bv, Vtb, 1);

  dim3 agrid(Bsz * Hsz, Ssz / 64);  // (64, 32)
  attn_k<<<agrid, 128, 0, stream>>>(Qb, Kb, Vtb, lens, ctx);

  gemm_bf16_k<<<ggrid, 256, 0, stream>>>(ctx, wob, bo, out, 2);
}
